// MultiHeadAttention_70239895159475
// MI455X (gfx1250) — compile-verified, matrix operand order checked by reading, not run on hardware
//
#include <hip/hip_runtime.h>

#define D_MODEL 512
#define NHEAD   8
#define DHEAD   64
#define BATCH   4
#define SEQ     2048
#define BS      (BATCH * SEQ)        // 8192
#define MINV    (-1e9f)

typedef __attribute__((ext_vector_type(16))) _Float16 v16h;
typedef __attribute__((ext_vector_type(8)))  _Float16 v8h;
typedef __attribute__((ext_vector_type(8)))  float    v8f;

// 16x32 f16 WMMA A/B fragment from a row-major f16 row pointer.
// Per-lane data = 8 contiguous halves at k0 and 8 at k0+16,
// where k0 = kbase + (lane>>4)*8. Both runs are 16B-aligned -> b128 loads
// (global_load_b128 from global, ds_load_b128 from LDS).
__device__ __forceinline__ v16h frag_from_f16(const _Float16* row, int k0) {
    v8h lo = *(const v8h*)(row + k0);
    v8h hi = *(const v8h*)(row + k0 + 16);
    return __builtin_shufflevector(lo, hi, 0,1,2,3,4,5,6,7,8,9,10,11,12,13,14,15);
}

// Same fragment but source is f32; convert on the fly (v_cvt_pk_f16_f32).
__device__ __forceinline__ v16h frag_from_f32(const float* __restrict__ row, int k0) {
    v16h f;
#pragma unroll
    for (int i = 0; i < 8; ++i) f[i]     = (_Float16)row[k0 + i];
#pragma unroll
    for (int i = 0; i < 8; ++i) f[8 + i] = (_Float16)row[k0 + 16 + i];
    return f;
}

#define WMMA_F16(a, b, c) \
    __builtin_amdgcn_wmma_f32_16x16x32_f16(false, (a), false, (b), (short)0, (c), false, false)

// ---------------------------------------------------------------------------
// Phase A: Q = query@Wq^T, K = keys@Wk^T (plain [BS,D] f16),
//          V = keys@Wv^T stored head-transposed Vt[B][H][dh][S] f16
//          (so the AV GEMM's B-fragments are contiguous row loads).
// Wave computes a 16x64 strip; grid.y selects which GEMM.
// ---------------------------------------------------------------------------
__global__ __launch_bounds__(256) void proj_kernel(
        const float* __restrict__ query, const float* __restrict__ keys,
        const float* __restrict__ Wq, const float* __restrict__ Wk,
        const float* __restrict__ Wv,
        _Float16* __restrict__ Qh, _Float16* __restrict__ Kh,
        _Float16* __restrict__ Vt) {
    const int wave = threadIdx.x >> 5;
    const int lane = threadIdx.x & 31;
    const int hi16 = lane >> 4;          // selects K-half of fragment
    const int mrow = lane & 15;          // A: M row / B: N col
    const int which = blockIdx.y;        // 0:Q 1:K 2:V
    const int g = blockIdx.x * 8 + wave; // 0..4095
    const int nstrip = g & 7;            // 8 strips of 64 cols
    const int mtile  = g >> 3;           // 0..511

    const float* X = (which == 0) ? query : keys;
    const float* W = (which == 0) ? Wq : (which == 1) ? Wk : Wv;

    const int m0 = mtile * 16;
    const int n0 = nstrip * 64;
    const float* arow = X + (size_t)(m0 + mrow) * D_MODEL;

    v8f acc[4] = {};
    for (int kk = 0; kk < D_MODEL; kk += 32) {
        const int k0 = kk + hi16 * 8;
        v16h a = frag_from_f32(arow, k0);
#pragma unroll
        for (int t = 0; t < 4; ++t) {
            // B[k][n] = W[n][k] : row n of W, contiguous in k
            v16h b = frag_from_f32(W + (size_t)(n0 + t * 16 + mrow) * D_MODEL, k0);
            acc[t] = WMMA_F16(a, b, acc[t]);
        }
    }

#pragma unroll
    for (int t = 0; t < 4; ++t) {
#pragma unroll
        for (int j = 0; j < 8; ++j) {
            const int m = m0 + hi16 * 8 + j;     // C: VGPR j -> M, lane hi -> +8
            const int n = n0 + t * 16 + mrow;    // C: lane -> N
            const _Float16 v = (_Float16)acc[t][j];
            if (which == 0) {
                Qh[(size_t)m * D_MODEL + n] = v;
            } else if (which == 1) {
                Kh[(size_t)m * D_MODEL + n] = v;
            } else {
                const int b = m >> 11, s = m & (SEQ - 1);
                const int h = n >> 6,  c = n & (DHEAD - 1);
                Vt[(((size_t)b * NHEAD + h) * DHEAD + c) * SEQ + s] = v;
            }
        }
    }
}

// ---------------------------------------------------------------------------
// Fused attention: per block = one (h*B+b, 16-query tile):
//   scores (WMMA) -> mask+scale -> softmax (shfl + LDS reductions)
//   -> write similarity once (non-temporal) -> stage P as f16 in LDS
//   -> AV (WMMA, A-fragments via ds_load_b128).
// Block = 512 threads = 16 waves; each wave owns 128 keys of the logit row.
// ---------------------------------------------------------------------------
__global__ __launch_bounds__(512) void attn_fused_kernel(
        const _Float16* __restrict__ Qh, const _Float16* __restrict__ Kh,
        const _Float16* __restrict__ Vt, const int* __restrict__ mask,
        float* __restrict__ sim, float* __restrict__ out) {
    // 16 rows x 1024 keys (one phase) of probabilities in f16.
    // Row stride 1032 halves = 2064B; 2064 % 256 = 16 -> the 16 A-fragment
    // rows land on distinct 16B bank groups (conflict-free ds_load_b128).
    __shared__ _Float16 Plds[16][1032];
    __shared__ float s_red[16][16];     // per-wave row stats
    __shared__ float s_rowmax[16];
    __shared__ float s_rowinv[16];
    __shared__ float outbuf[16][64];    // AV partial combine

    const int tid  = threadIdx.x;
    const int wave = tid >> 5;           // 0..15
    const int lane = tid & 31;
    const int hi16 = lane >> 4;
    const int mrow = lane & 15;
    const int hb = blockIdx.y;           // h*B + b
    const int h  = hb >> 2, b = hb & 3;
    const int q0 = blockIdx.x * 16;

    // zero combine buffer (consumed only after several barriers)
    for (int i = tid; i < 16 * 64; i += 512) ((float*)outbuf)[i] = 0.f;

    // ---- scores: this wave handles keys [wave*128, wave*128+128)
    const _Float16* qrow = Qh + (size_t)(b * SEQ + q0 + mrow) * D_MODEL + h * DHEAD;
    const v16h a0 = frag_from_f16(qrow, hi16 * 8);
    const v16h a1 = frag_from_f16(qrow, 32 + hi16 * 8);

    v8f acc[8];
#pragma unroll
    for (int t = 0; t < 8; ++t) {
        const int n = wave * 128 + t * 16 + mrow;
        const _Float16* krow = Kh + (size_t)(b * SEQ + n) * D_MODEL + h * DHEAD;
        v16h b0 = frag_from_f16(krow, hi16 * 8);
        v16h b1 = frag_from_f16(krow, 32 + hi16 * 8);
        v8f c = {};
        c = WMMA_F16(a0, b0, c);
        c = WMMA_F16(a1, b1, c);
        acc[t] = c;
    }

    // ---- mask + scale (in registers)
    const float scale = 0.04419417382415922f;  // 1/sqrt(512)
#pragma unroll
    for (int t = 0; t < 8; ++t) {
        const int n = wave * 128 + t * 16 + mrow;
#pragma unroll
        for (int j = 0; j < 8; ++j) {
            const int q = q0 + hi16 * 8 + j;
            const int mv = mask[((size_t)b * SEQ + q) * SEQ + n];
            acc[t][j] = (mv > 0) ? acc[t][j] * scale : MINV;
        }
    }

    // ---- row max: per-lane over t, then shfl within the 16-lane group,
    //      then LDS across the 16 waves.
#pragma unroll
    for (int j = 0; j < 8; ++j) {
        float m = acc[0][j];
#pragma unroll
        for (int t = 1; t < 8; ++t) m = fmaxf(m, acc[t][j]);
#pragma unroll
        for (int s = 1; s <= 8; s <<= 1) m = fmaxf(m, __shfl_xor(m, s));
        if (mrow == 0) s_red[wave][hi16 * 8 + j] = m;
    }
    __syncthreads();
    if (tid < 16) {
        float m = s_red[0][tid];
        for (int w2 = 1; w2 < 16; ++w2) m = fmaxf(m, s_red[w2][tid]);
        s_rowmax[tid] = m;
    }
    __syncthreads();

    // ---- exp + row sum
#pragma unroll
    for (int j = 0; j < 8; ++j) {
        const float rm = s_rowmax[hi16 * 8 + j];
        float s = 0.f;
#pragma unroll
        for (int t = 0; t < 8; ++t) {
            const float e = __expf(acc[t][j] - rm);
            acc[t][j] = e;
            s += e;
        }
#pragma unroll
        for (int sh = 1; sh <= 8; sh <<= 1) s += __shfl_xor(s, sh);
        if (mrow == 0) s_red[wave][hi16 * 8 + j] = s;
    }
    __syncthreads();
    if (tid < 16) {
        float s = 0.f;
        for (int w2 = 0; w2 < 16; ++w2) s += s_red[w2][tid];
        s_rowinv[tid] = 1.0f / s;
    }
    __syncthreads();

    // ---- normalize; write similarity exactly once, non-temporal
    //      (write-once data: keep L2 for mask/K/V reuse)
#pragma unroll
    for (int j = 0; j < 8; ++j) {
        const int q = q0 + hi16 * 8 + j;
        const float inv = s_rowinv[hi16 * 8 + j];
        float* srow = sim + ((size_t)hb * SEQ + q) * SEQ;
#pragma unroll
        for (int t = 0; t < 8; ++t) {
            const int n = wave * 128 + t * 16 + mrow;
            const float p = acc[t][j] * inv;
            acc[t][j] = p;
            __builtin_nontemporal_store(p, srow + n);
        }
    }

    // ---- AV with P staged in LDS, two key-phases of 1024 keys.
    // Per phase: wave -> (dh slab ctile, 256-key segment kseg).
    const int ctile = wave & 3;
    const int kseg  = wave >> 2;
    const _Float16* vrow =
        Vt + (((size_t)b * NHEAD + h) * DHEAD + ctile * 16 + mrow) * SEQ;

    v8f vacc = {};
    for (int ph = 0; ph < 2; ++ph) {
        __syncthreads();  // everyone done reading previous P contents
        if ((wave >> 3) == ph) {
            const int wloc = wave & 7;   // key offset within this phase
#pragma unroll
            for (int t = 0; t < 8; ++t) {
#pragma unroll
                for (int j = 0; j < 8; ++j) {
                    Plds[hi16 * 8 + j][wloc * 128 + t * 16 + mrow] =
                        (_Float16)acc[t][j];
                }
            }
        }
        __syncthreads();
#pragma unroll
        for (int ks = 0; ks < 8; ++ks) {
            const int kl0 = kseg * 256 + ks * 32 + hi16 * 8;
            v16h pa = frag_from_f16(&Plds[mrow][0], kl0);       // ds_load_b128
            v16h vb = frag_from_f16(vrow, ph * 1024 + kl0);     // global b128
            vacc = WMMA_F16(pa, vb, vacc);
        }
    }
    __syncthreads();

    // ---- combine the 4 key-segment partials per dh slab (ds_add_f32)
#pragma unroll
    for (int j = 0; j < 8; ++j)
        atomicAdd(&outbuf[hi16 * 8 + j][ctile * 16 + mrow], vacc[j]);
    __syncthreads();

    // ---- write output tile: 16 x 64 values, 2 per thread (non-temporal)
#pragma unroll
    for (int i = 0; i < 2; ++i) {
        const int idx = tid * 2 + i;
        const int ql = idx >> 6, c = idx & 63;
        __builtin_nontemporal_store(
            outbuf[ql][c],
            out + ((size_t)b * SEQ + q0 + ql) * D_MODEL + h * DHEAD + c);
    }
}

// ---------------------------------------------------------------------------
extern "C" void kernel_launch(void* const* d_in, const int* in_sizes, int n_in,
                              void* d_out, int out_size, void* d_ws, size_t ws_size,
                              hipStream_t stream) {
    const float* query = (const float*)d_in[0];
    const float* keys  = (const float*)d_in[1];
    const int*   mask  = (const int*)d_in[2];
    const float* Wq    = (const float*)d_in[3];
    const float* Wk    = (const float*)d_in[4];
    const float* Wv    = (const float*)d_in[5];

    float* out = (float*)d_out;                              // [B,S,D]
    float* sim = out + (size_t)BATCH * SEQ * D_MODEL;        // [H*B,S,S]

    _Float16* Qh = (_Float16*)d_ws;                          // [BS, D]
    _Float16* Kh = Qh + (size_t)BS * D_MODEL;                // [BS, D]
    _Float16* Vt = Kh + (size_t)BS * D_MODEL;                // [B,H,dh,S]

    proj_kernel<<<dim3(512, 3), 256, 0, stream>>>(query, keys, Wq, Wk, Wv,
                                                  Qh, Kh, Vt);
    attn_fused_kernel<<<dim3(SEQ / 16, NHEAD * BATCH), 512, 0, stream>>>(
        Qh, Kh, Vt, mask, sim, out);
}
